// SelfAttn_17351667876247
// MI455X (gfx1250) — matrix operand order repaired and checked by reading, not compiled, not run on hardware
//
#include <hip/hip_runtime.h>

typedef _Float16 half_t;
typedef __attribute__((ext_vector_type(16))) _Float16 v16h;
typedef __attribute__((ext_vector_type(8)))  _Float16 v8h;
typedef __attribute__((ext_vector_type(8)))  float    v8f;
typedef unsigned int u32x4 __attribute__((ext_vector_type(4)));
typedef int          i32x4 __attribute__((ext_vector_type(4)));
typedef int          i32x8 __attribute__((ext_vector_type(8)));

#define BATCH 8
#define SEQ   2048
#define DIM   256
#define NH    8
#define DH    32
#define ROWS  (BATCH*SEQ)   // 16384
#define NCOLS1 1024         // fused [q|k|nonatt|v] gemm columns

#define KSTRIDE 48   // halfs: 32 payload + 16 pad (TDM: 16 dw + 8 dw pad)
#define VSTRIDE 80   // halfs: 64 payload + 16 pad (TDM: 32 dw + 8 dw pad)
#define PSTRIDE 72   // halfs, padded (16 q x 64 keys) per wave

#if __has_builtin(__builtin_amdgcn_tensor_load_to_lds)
#define HAVE_TDM 1
#else
#define HAVE_TDM 0
#endif

__device__ __forceinline__ v16h cat16(v8h lo, v8h hi) {
  return __builtin_shufflevector(lo, hi, 0,1,2,3,4,5,6,7,8,9,10,11,12,13,14,15);
}
__device__ __forceinline__ v8f wmma16(v16h a, v16h b, v8f c) {
  return __builtin_amdgcn_wmma_f32_16x16x32_f16(false, a, false, b, (short)0, c, false, false);
}

#if HAVE_TDM
// 2-D tile DMA: global (row-major, d0_stride elems/row, 2B elems) -> LDS,
// with per-row LDS padding (pad codes per D# group1 encoding).
__device__ __forceinline__ void tdm_load_2d(
    unsigned lds_byte_addr, const void* gptr,
    unsigned tensor_d0, unsigned tensor_d1,
    unsigned tile_d0, unsigned tile_d1,
    unsigned d0_stride, unsigned pad_int_code, unsigned pad_amt_code) {
  unsigned long long ga = (unsigned long long)(size_t)gptr;
  u32x4 g0;
  g0.x = 1u;                                     // count=1, user mode
  g0.y = lds_byte_addr;                          // bits 63:32
  g0.z = (unsigned)ga;                           // global_addr[31:0]
  g0.w = (unsigned)(ga >> 32) | 0x80000000u;     // addr[56:32] | type=2
  i32x8 g1;
  g1[0] = (int)((1u << 16)                       // data_size = 2B
              | (1u << 20)                       // pad_enable
              | (pad_int_code << 22)
              | (pad_amt_code << 25));
  g1[1] = (int)((tensor_d0 & 0xFFFFu) << 16);                               // dim0[15:0]
  g1[2] = (int)(((tensor_d0 >> 16) & 0xFFFFu) | ((tensor_d1 & 0xFFFFu) << 16));
  g1[3] = (int)(((tensor_d1 >> 16) & 0xFFFFu) | ((tile_d0 & 0xFFFFu) << 16)); // tile_dim0
  g1[4] = (int)(tile_d1 & 0xFFFFu);              // tile_dim1 (tile_dim2 = 0)
  g1[5] = (int)d0_stride;                        // tensor_dim0_stride[31:0]
  g1[6] = 0;
  g1[7] = 0;
  i32x4 gz = {0, 0, 0, 0};
#if defined(__clang_major__) && (__clang_major__ >= 23)
  i32x8 gz8 = {0, 0, 0, 0, 0, 0, 0, 0};
  __builtin_amdgcn_tensor_load_to_lds(g0, g1, gz, gz, gz8, 0);
#else
  __builtin_amdgcn_tensor_load_to_lds(g0, g1, gz, gz, 0);
#endif
}
#endif

// ---------------- Kernel 1: f32 -> f16 conversions + weight transposes ----
__global__ void __launch_bounds__(256) cvt_kernel(
    const float* __restrict__ x, const float* __restrict__ w1,
    const float* __restrict__ wv, const float* __restrict__ wo,
    half_t* __restrict__ xh, half_t* __restrict__ wcatT, half_t* __restrict__ woT) {
  const int NX = ROWS*DIM;
  const int NW = NCOLS1*DIM;
  const int NO = DIM*DIM;
  int idx = blockIdx.x*256 + threadIdx.x;
  if (idx < NX) {
    xh[idx] = (half_t)x[idx];
  } else if (idx < NX + NW) {
    int t = idx - NX; int c = t >> 8; int r = t & 255;
    float v = (c < 768) ? w1[r*768 + c] : wv[r*256 + (c - 768)];
    wcatT[c*256 + r] = (half_t)v;
  } else if (idx < NX + NW + NO) {
    int t = idx - NX - NW; int c = t >> 8; int r = t & 255;
    woT[c*256 + r] = (half_t)wo[r*256 + c];
  }
}

// ---------------- Kernel 2: fused QKV / non-att projection GEMM -----------
// C[16384 x 1024] = Xh @ Wcat; 64-row x 256-col block, 2 M-tiles per wave,
// register-level software pipeline on the K loop.
__global__ void __launch_bounds__(256) qkv_kernel(
    const half_t* __restrict__ xh, const half_t* __restrict__ wcatT,
    const float* __restrict__ b1, const float* __restrict__ bv,
    const unsigned char* __restrict__ mask,
    half_t* __restrict__ qh, half_t* __restrict__ kh,
    float* __restrict__ nonatt, half_t* __restrict__ vT) {
  const int lane = threadIdx.x & 31;
  const int w    = threadIdx.x >> 5;
  const int wm = w & 1, wn = w >> 1;
  const int row0 = blockIdx.x*64 + wm*32;
  const int colbase = blockIdx.y*256 + wn*64;
  const int l15 = lane & 15, hb = lane >> 4;

  v8f acc[2][4];
  #pragma unroll
  for (int m = 0; m < 2; m++)
    #pragma unroll
    for (int t = 0; t < 4; t++) acc[m][t] = (v8f){0,0,0,0,0,0,0,0};

  const half_t* ap0 = xh + (row0 + l15)*DIM;
  const half_t* ap1 = xh + (row0 + 16 + l15)*DIM;

  v16h a[2], bcur[4];
  a[0] = *(const v16h*)(ap0 + hb*16);
  a[1] = *(const v16h*)(ap1 + hb*16);
  #pragma unroll
  for (int t = 0; t < 4; t++)
    bcur[t] = *(const v16h*)(wcatT + (colbase + t*16 + l15)*DIM + hb*16);

  #pragma unroll
  for (int k = 0; k < DIM; k += 32) {
    v16h an[2] = {a[0], a[1]};
    v16h bn[4] = {bcur[0], bcur[1], bcur[2], bcur[3]};
    if (k + 32 < DIM) {  // prefetch next stage (compile-time on unroll)
      an[0] = *(const v16h*)(ap0 + k + 32 + hb*16);
      an[1] = *(const v16h*)(ap1 + k + 32 + hb*16);
      #pragma unroll
      for (int t = 0; t < 4; t++)
        bn[t] = *(const v16h*)(wcatT + (colbase + t*16 + l15)*DIM + k + 32 + hb*16);
    }
    #pragma unroll
    for (int t = 0; t < 4; t++) {
      acc[0][t] = wmma16(a[0], bcur[t], acc[0][t]);
      acc[1][t] = wmma16(a[1], bcur[t], acc[1][t]);
    }
    a[0] = an[0]; a[1] = an[1];
    #pragma unroll
    for (int t = 0; t < 4; t++) bcur[t] = bn[t];
  }

  #pragma unroll
  for (int m = 0; m < 2; m++) {
    int row0m = row0 + m*16;
    #pragma unroll
    for (int t = 0; t < 4; t++) {
      int colb = colbase + t*16;
      int col  = colb + l15;
      float bias = (col < 768) ? b1[col] : bv[col - 768];
      #pragma unroll
      for (int i = 0; i < 8; i++) {
        int row = row0m + i + hb*8;
        float mv = mask[row] ? 1.0f : 0.0f;
        float v = (acc[m][t][i] + bias) * mv;
        if (colb < 256) {
          qh[row*DIM + col] = (half_t)(v * 0.17677669529663689f);  // fold 1/sqrt(32)
        } else if (colb < 512) {
          kh[row*DIM + (col - 256)] = (half_t)v;
        } else if (colb < 768) {
          nonatt[row*DIM + (col - 512)] = v;
        } else {
          int c = col - 768;
          int hh = c >> 5, dv = c & 31;
          int bi = row >> 11, n = row & 2047;
          vT[((bi*NH + hh)*DH + dv)*SEQ + n] = (half_t)v;          // key-minor V
        }
      }
    }
  }
}

// ---------------- Kernel 3: flash attention, TDM double-buffered K/V ------
__global__ void __launch_bounds__(256) attn_kernel(
    const half_t* __restrict__ qh, const half_t* __restrict__ kh,
    const half_t* __restrict__ vT, const unsigned char* __restrict__ mask,
    half_t* __restrict__ atth) {
  __shared__ __align__(32) half_t k_lds[2][64*KSTRIDE];
  __shared__ __align__(32) half_t v_lds[2][DH*VSTRIDE];
  __shared__ __align__(32) half_t p_lds[8*16*PSTRIDE];

  const int lane = threadIdx.x & 31;
  const int w    = threadIdx.x >> 5;
  const int l15 = lane & 15, hb = lane >> 4;
  const int bh = blockIdx.x;
  const int b = bh >> 3, h = bh & 7;
  const int qbase = blockIdx.y*128 + w*16;

  // Q fragment: 16 queries x 32 d (A layout, loaded once)
  const int qoff = (b*SEQ + qbase + l15)*DIM + h*DH;
  v16h aQ = cat16(*(const v8h*)(qh + qoff + hb*16),
                  *(const v8h*)(qh + qoff + hb*16 + 8));

  v8f o0 = (v8f){0,0,0,0,0,0,0,0};
  v8f o1 = (v8f){0,0,0,0,0,0,0,0};
  float mrow[8], lrow[8];
  #pragma unroll
  for (int i = 0; i < 8; i++) { mrow[i] = -3.0e38f; lrow[i] = 0.0f; }

  half_t* pw = p_lds + w*16*PSTRIDE;
  const v8f vzero = (v8f){0,0,0,0,0,0,0,0};
  const half_t* kbase = kh + (size_t)(b*SEQ)*DIM + h*DH;
  const half_t* vbase = vT + (size_t)((b*NH + h)*DH)*SEQ;

#if HAVE_TDM
  if (w == 0) { // prologue: DMA block 0 into buffer 0
    tdm_load_2d((unsigned)(size_t)&k_lds[0][0], kbase, 256, SEQ, 32, 64, 256, 3, 7);
    tdm_load_2d((unsigned)(size_t)&v_lds[0][0], vbase, SEQ, DH, 64, DH, SEQ, 4, 7);
  }
#endif

  int ibuf = 0;
  for (int kblk = 0; kblk < SEQ; kblk += 64, ibuf ^= 1) {
#if HAVE_TDM
    __builtin_amdgcn_s_wait_tensorcnt(0);   // no-op for waves that issued nothing
    __syncthreads();                        // publish buffer ibuf to all waves
    if (w == 0 && kblk + 64 < SEQ) {        // DMA next block into other buffer
      tdm_load_2d((unsigned)(size_t)&k_lds[ibuf ^ 1][0],
                  kbase + (size_t)(kblk + 64)*DIM, 256, SEQ, 32, 64, 256, 3, 7);
      tdm_load_2d((unsigned)(size_t)&v_lds[ibuf ^ 1][0],
                  vbase + (kblk + 64), SEQ, DH, 64, DH, SEQ, 4, 7);
    }
#else
    __syncthreads();
    { // synchronous cooperative fill fallback
      int krow = threadIdx.x >> 2, kseg = threadIdx.x & 3;
      *(v8h*)(&k_lds[ibuf][0] + krow*KSTRIDE + kseg*8) =
          *(const v8h*)(kbase + (size_t)(kblk + krow)*DIM + kseg*8);
      int dvr = threadIdx.x >> 3, vseg = threadIdx.x & 7;
      *(v8h*)(&v_lds[ibuf][0] + dvr*VSTRIDE + vseg*8) =
          *(const v8h*)(vbase + (size_t)dvr*SEQ + kblk + vseg*8);
    }
    __syncthreads();
#endif

    // scores: 4 tiles of 16q x 16k, K-dim = 32
    float s[4][8];
    #pragma unroll
    for (int t = 0; t < 4; t++) {
      v16h bK = *(const v16h*)(&k_lds[ibuf][0] + (t*16 + l15)*KSTRIDE + hb*16);
      v8f sf = wmma16(aQ, bK, vzero);
      int key = kblk + t*16 + l15;
      bool mk = mask[b*SEQ + key] != 0;
      #pragma unroll
      for (int i = 0; i < 8; i++) s[t][i] = mk ? sf[i] : -1.0e9f;
    }

    // online softmax (row = 16 lanes of a half-wave)
    float p[4][8];
    #pragma unroll
    for (int i = 0; i < 8; i++) {
      float vmax = fmaxf(fmaxf(s[0][i], s[1][i]), fmaxf(s[2][i], s[3][i]));
      #pragma unroll
      for (int off = 1; off < 16; off <<= 1) vmax = fmaxf(vmax, __shfl_xor(vmax, off, 32));
      float mnew = fmaxf(mrow[i], vmax);
      float c = __expf(mrow[i] - mnew);
      float rs = 0.0f;
      #pragma unroll
      for (int t = 0; t < 4; t++) { p[t][i] = __expf(s[t][i] - mnew); rs += p[t][i]; }
      #pragma unroll
      for (int off = 1; off < 16; off <<= 1) rs += __shfl_xor(rs, off, 32);
      lrow[i] = lrow[i]*c + rs;
      mrow[i] = mnew;
      o0[i] *= c; o1[i] *= c;
    }

    // P: C-layout regs -> per-wave LDS (f16) -> A-layout frags
    #pragma unroll
    for (int t = 0; t < 4; t++)
      #pragma unroll
      for (int i = 0; i < 8; i++)
        pw[(i + hb*8)*PSTRIDE + t*16 + l15] = (half_t)p[t][i];
    asm volatile("s_wait_dscnt 0" ::: "memory");  // DS in-order within wave

    // O += P(16x64) @ V(64x32)
    #pragma unroll
    for (int kc = 0; kc < 64; kc += 32) {
      v16h aP = cat16(*(const v8h*)(pw + l15*PSTRIDE + kc + hb*16),
                      *(const v8h*)(pw + l15*PSTRIDE + kc + hb*16 + 8));
      v16h bV0 = *(const v16h*)(&v_lds[ibuf][0] + (l15)*VSTRIDE + kc + hb*16);
      v16h bV1 = *(const v16h*)(&v_lds[ibuf][0] + (16 + l15)*VSTRIDE + kc + hb*16);
      o0 = wmma16(aP, bV0, o0);
      o1 = wmma16(aP, bV1, o1);
    }
  }

  #pragma unroll
  for (int i = 0; i < 8; i++) {
    float inv = 1.0f / lrow[i];
    int row = qbase + i + hb*8;
    int base = (b*SEQ + row)*DIM + h*DH;
    atth[base + l15]      = (half_t)(o0[i]*inv);
    atth[base + 16 + l15] = (half_t)(o1[i]*inv);
  }
}

// ---------------- Kernel 4: output projection + residual ------------------
__global__ void __launch_bounds__(256) out_kernel(
    const half_t* __restrict__ atth, const half_t* __restrict__ woT,
    const float* __restrict__ bo, const float* __restrict__ nonatt,
    const unsigned char* __restrict__ mask, float* __restrict__ out) {
  const int lane = threadIdx.x & 31;
  const int w    = threadIdx.x >> 5;
  const int wm = w & 1, wn = w >> 1;
  const int row0 = blockIdx.x*64 + wm*32;
  const int colbase = wn*64;
  const int l15 = lane & 15, hb = lane >> 4;

  v8f acc[2][4];
  #pragma unroll
  for (int m = 0; m < 2; m++)
    #pragma unroll
    for (int t = 0; t < 4; t++) acc[m][t] = (v8f){0,0,0,0,0,0,0,0};

  const half_t* ap0 = atth + (row0 + l15)*DIM;
  const half_t* ap1 = atth + (row0 + 16 + l15)*DIM;

  v16h a[2], bcur[4];
  a[0] = *(const v16h*)(ap0 + hb*16);
  a[1] = *(const v16h*)(ap1 + hb*16);
  #pragma unroll
  for (int t = 0; t < 4; t++)
    bcur[t] = *(const v16h*)(woT + (colbase + t*16 + l15)*DIM + hb*16);

  #pragma unroll
  for (int k = 0; k < DIM; k += 32) {
    v16h an[2] = {a[0], a[1]};
    v16h bn[4] = {bcur[0], bcur[1], bcur[2], bcur[3]};
    if (k + 32 < DIM) {
      an[0] = *(const v16h*)(ap0 + k + 32 + hb*16);
      an[1] = *(const v16h*)(ap1 + k + 32 + hb*16);
      #pragma unroll
      for (int t = 0; t < 4; t++)
        bn[t] = *(const v16h*)(woT + (colbase + t*16 + l15)*DIM + k + 32 + hb*16);
    }
    #pragma unroll
    for (int t = 0; t < 4; t++) {
      acc[0][t] = wmma16(a[0], bcur[t], acc[0][t]);
      acc[1][t] = wmma16(a[1], bcur[t], acc[1][t]);
    }
    a[0] = an[0]; a[1] = an[1];
    #pragma unroll
    for (int t = 0; t < 4; t++) bcur[t] = bn[t];
  }

  #pragma unroll
  for (int m = 0; m < 2; m++) {
    int row0m = row0 + m*16;
    #pragma unroll
    for (int t = 0; t < 4; t++) {
      int col = colbase + t*16 + l15;
      float bias = bo[col];
      #pragma unroll
      for (int i = 0; i < 8; i++) {
        int row = row0m + i + hb*8;
        float mv = mask[row] ? 1.0f : 0.0f;
        out[row*DIM + col] = (nonatt[row*DIM + col] + acc[m][t][i] + bias) * mv;
      }
    }
  }
}

// ---------------- Host launch ---------------------------------------------
extern "C" void kernel_launch(void* const* d_in, const int* in_sizes, int n_in,
                              void* d_out, int out_size, void* d_ws, size_t ws_size,
                              hipStream_t stream) {
  const float* x  = (const float*)d_in[0];
  const float* w1 = (const float*)d_in[1];
  const float* b1 = (const float*)d_in[2];
  const float* wv = (const float*)d_in[3];
  const float* bv = (const float*)d_in[4];
  const float* wo = (const float*)d_in[5];
  const float* bo = (const float*)d_in[6];
  const unsigned char* mask = (const unsigned char*)d_in[7];
  float* out = (float*)d_out;

  char* ws = (char*)d_ws;
  half_t* xh     = (half_t*)(ws);                 //  8,388,608 B
  half_t* qh     = (half_t*)(ws + 8388608);       //  8,388,608 B
  half_t* kh     = (half_t*)(ws + 16777216);      //  8,388,608 B
  half_t* vT     = (half_t*)(ws + 25165824);      //  8,388,608 B
  half_t* atth   = (half_t*)(ws + 33554432);      //  8,388,608 B
  half_t* wcatT  = (half_t*)(ws + 41943040);      //    524,288 B
  half_t* woT    = (half_t*)(ws + 42467328);      //    131,072 B
  float*  nonatt = (float*)(ws + 42598400);       // 16,777,216 B  (total ~59.4 MB)

  int total = ROWS*DIM + NCOLS1*DIM + DIM*DIM;
  cvt_kernel<<<(total + 255)/256, 256, 0, stream>>>(x, w1, wv, wo, xh, wcatT, woT);
  qkv_kernel<<<dim3(ROWS/64, 4), 256, 0, stream>>>(xh, wcatT, b1, bv, mask, qh, kh, nonatt, vT);
  attn_kernel<<<dim3(BATCH*NH, SEQ/128), 256, 0, stream>>>(qh, kh, vT, mask, atth);
  out_kernel<<<dim3(ROWS/64), 256, 0, stream>>>(atth, woT, bo, nonatt, mask, out);
}
